// StaticBoardGCN_82557861363947
// MI455X (gfx1250) — compile-verified, matrix operand order checked by reading, not run on hardware
//
#include <hip/hip_runtime.h>
#include <math.h>

#define N_VERTS 54
#define VP 64          // padded vertex count
#define NODE_DIM 16
#define HID 128
#define APAD 72        // LDS K-pitch for A_hat (bf16), 144 B (16B aligned)
#define HPAD 136       // LDS pitch for H rows / W k-dim (bf16), 272 B (16B aligned)
#define KP   72        // LDS K-pitch for transposed T (bf16), 144 B (16B aligned)
#define TPAD 132       // LDS pitch for f32 mix output
#define EPSV 1e-5f

typedef __attribute__((ext_vector_type(16))) __bf16 bf16x16;
typedef __attribute__((ext_vector_type(8)))  __bf16 bf16x8;
typedef __attribute__((ext_vector_type(8)))  float  v8f;

// 16x32 bf16 fragment from a row-major [major][ld] LDS buffer, per ISA 7.12.2:
// lanes 0-15 hold rows, K=0..7 & 16..23; lanes 16-31 same rows, K=8..15 & 24..31.
// Both 8-element groups are 16B-contiguous -> two ds_load_b128.
__device__ __forceinline__ bf16x16 frag_rm(const __bf16* base, int ld, int major, int kb, int lane) {
    int row = major + (lane & 15);
    int kh  = (lane >> 4) << 3;
    const bf16x8* p = reinterpret_cast<const bf16x8*>(base + row * ld + kb + kh);
    union { bf16x16 v; bf16x8 h[2]; } u;
    u.h[0] = p[0];     // K = kb+kh .. kb+kh+7
    u.h[1] = p[2];     // K = kb+16+kh .. kb+16+kh+7
    return u.v;
}

__global__ __launch_bounds__(256, 1)
void gcn_fused_kernel(const float* __restrict__ X, const float* __restrict__ A_hat,
                      const int* __restrict__ v1_idx, const int* __restrict__ v2_idx,
                      const int* __restrict__ seat,
                      const float* conv0_W, const float* conv0_b, const float* ln0_g, const float* ln0_b,
                      const float* conv1_W, const float* conv1_b, const float* ln1_g, const float* ln1_b,
                      const float* conv2_W, const float* conv2_b, const float* ln2_g, const float* ln2_b,
                      const float* seat_emb,
                      const float* mlp_W1, const float* mlp_b1, const float* mlp_W2, const float* mlp_b2,
                      const float* mlp_W3, const float* mlp_b3,
                      const float* rank_W1, const float* rank_b1, const float* rank_W2, const float* rank_b2,
                      float* __restrict__ out, int n_total)
{
    __shared__ __attribute__((aligned(16))) __bf16 sA[VP][APAD];    // padded A_hat
    __shared__ __attribute__((aligned(16))) __bf16 sH[VP][HPAD];    // layer input H (row-major)
    __shared__ __attribute__((aligned(16))) __bf16 sBt[HID][KP];    // T = H*W+b, TRANSPOSED [col][k]
    __shared__ __attribute__((aligned(16))) unsigned char sUraw[HID * HPAD * 2]; // Wt (bf16) / mix out (f32)
    __shared__ float sBias[HID];
    __shared__ float sHvec[400];
    __shared__ float sZ1[HID];
    __shared__ float sR[64];
    __shared__ float sZ2[64];
    __shared__ float sRed[8];

    __bf16 (*sWt)[HPAD] = reinterpret_cast<__bf16 (*)[HPAD]>(sUraw); // W TRANSPOSED [col][k]
    float  (*sT)[TPAD]  = reinterpret_cast<float  (*)[TPAD]>(sUraw); // f32 mix output (row-major)

    const int tid  = threadIdx.x;
    const int lane = tid & 31;
    const int wave = tid >> 5;
    const int n    = blockIdx.x;

    // ---- stage A_hat (zero-padded 64x64) and X (zero-padded 64x32) ----
    for (int i = tid; i < VP * VP; i += 256) {
        int r = i >> 6, c = i & 63;
        float val = (r < N_VERTS && c < N_VERTS) ? A_hat[r * N_VERTS + c] : 0.f;
        sA[r][c] = (__bf16)val;
    }
    for (int i = tid; i < VP * 32; i += 256) {
        int r = i >> 5, c = i & 31;
        float val = (r < N_VERTS && c < NODE_DIM) ? X[((size_t)n * N_VERTS + r) * NODE_DIM + c] : 0.f;
        sH[r][c] = (__bf16)val;
    }

    const float* Wl[3]  = {conv0_W, conv1_W, conv2_W};
    const float* bl[3]  = {conv0_b, conv1_b, conv2_b};
    const float* gl[3]  = {ln0_g, ln1_g, ln2_g};
    const float* bel[3] = {ln0_b, ln1_b, ln2_b};
    const int    Kl[3]  = {32, HID, HID};

    // tile assignment: wave -> (row block, column half); 4 col tiles per wave
    const int rb   = wave >> 1;
    const int ch   = wave & 1;
    const int mr   = rb * 16;
    const int rowD = (lane >> 4) << 3;   // 0 or 8 (C/D layout)
    const int colD = lane & 15;

    for (int layer = 0; layer < 3; ++layer) {
        const int K   = Kl[layer];
        const int Din = (layer == 0) ? NODE_DIM : HID;

        // load weight transposed into LDS (bf16), zero-pad K; load bias
        for (int i = tid; i < K * HID; i += 256) {
            int k = i >> 7, c = i & 127;
            float val = (k < Din) ? Wl[layer][k * HID + c] : 0.f;
            sWt[c][k] = (__bf16)val;
        }
        if (tid < HID) sBias[tid] = bl[layer][tid];
        __syncthreads();

        // prefetch next layer's weights into cache while we compute
        if (layer < 2) __builtin_prefetch(Wl[layer + 1] + (tid << 6), 0, 0);

        // ---- GEMM1: T = H * W  (64 x K)*(K x 128), bf16 WMMA, f32 acc ----
        {
            v8f acc[4];
#pragma unroll
            for (int c = 0; c < 4; ++c)
#pragma unroll
                for (int e = 0; e < 8; ++e) acc[c][e] = 0.f;
            for (int kb = 0; kb < K; kb += 32) {
                bf16x16 a = frag_rm(&sH[0][0], HPAD, mr, kb, lane);
#pragma unroll
                for (int c = 0; c < 4; ++c) {
                    bf16x16 b = frag_rm(&sWt[0][0], HPAD, (ch * 4 + c) * 16, kb, lane);
                    acc[c] = __builtin_amdgcn_wmma_f32_16x16x32_bf16(
                        false, a, false, b, (short)0, acc[c], false, false);
                }
            }
            // epilogue: +bias, cvt to bf16, one packed 16B store per tile (transposed layout)
#pragma unroll
            for (int c = 0; c < 4; ++c) {
                int col = (ch * 4 + c) * 16 + colD;
                float bias = sBias[col];
                bf16x8 pk;
#pragma unroll
                for (int vv = 0; vv < 8; ++vv)
                    pk[vv] = (__bf16)(acc[c][vv] + bias);
                *reinterpret_cast<bf16x8*>(&sBt[col][mr + rowD]) = pk;
            }
        }
        __syncthreads();

        // ---- GEMM2: M = A_hat * T  (64 x 64)*(64 x 128), f32 out to sT ----
        {
            v8f acc[4];
#pragma unroll
            for (int c = 0; c < 4; ++c)
#pragma unroll
                for (int e = 0; e < 8; ++e) acc[c][e] = 0.f;
#pragma unroll
            for (int kb = 0; kb < VP; kb += 32) {
                bf16x16 a = frag_rm(&sA[0][0], APAD, mr, kb, lane);
#pragma unroll
                for (int c = 0; c < 4; ++c) {
                    bf16x16 b = frag_rm(&sBt[0][0], KP, (ch * 4 + c) * 16, kb, lane);
                    acc[c] = __builtin_amdgcn_wmma_f32_16x16x32_bf16(
                        false, a, false, b, (short)0, acc[c], false, false);
                }
            }
#pragma unroll
            for (int c = 0; c < 4; ++c) {
                int col = (ch * 4 + c) * 16 + colD;
#pragma unroll
                for (int vv = 0; vv < 8; ++vv)
                    sT[mr + rowD + vv][col] = acc[c][vv];
            }
        }
        __syncthreads();

        // ---- LayerNorm + ReLU + residual (wave-per-8-rows, shfl reductions) ----
        {
            const float* g  = gl[layer];
            const float* be = bel[layer];
            for (int r8 = 0; r8 < 8; ++r8) {
                int row = wave * 8 + r8;
                float x0 = sT[row][lane], x1 = sT[row][lane + 32];
                float x2 = sT[row][lane + 64], x3 = sT[row][lane + 96];
                float s = x0 + x1 + x2 + x3;
                float q = x0 * x0 + x1 * x1 + x2 * x2 + x3 * x3;
#pragma unroll
                for (int off = 16; off > 0; off >>= 1) {
                    s += __shfl_xor(s, off, 32);
                    q += __shfl_xor(q, off, 32);
                }
                float mean = s * (1.f / HID);
                float var  = q * (1.f / HID) - mean * mean;
                float rs   = rsqrtf(var + EPSV);
                float xs[4] = {x0, x1, x2, x3};
#pragma unroll
                for (int j = 0; j < 4; ++j) {
                    int c = lane + 32 * j;
                    float y = (xs[j] - mean) * rs * g[c] + be[c];
                    y = fmaxf(y, 0.f);
                    if (layer > 0) y += (float)sH[row][c];
                    sH[row][c] = (__bf16)y;
                }
            }
        }
        __syncthreads();
    }

    // ---- heads: gather v1/v2, mean-pool, seat embedding, two small MLPs ----
    int v1 = v1_idx[n], v2 = v2_idx[n];
    int st = seat[n]; st = st < 0 ? 0 : (st > 3 ? 3 : st);
    if (tid < HID) {
        int c = tid;
        sHvec[c]           = (float)sH[v1][c];
        sHvec[HID + c]     = (float)sH[v2][c];
        float acc = 0.f;
        for (int vtx = 0; vtx < N_VERTS; ++vtx) acc += (float)sH[vtx][c];
        sHvec[2 * HID + c] = acc * (1.f / N_VERTS);
    } else if (tid < HID + 8) {
        int t = tid - HID;
        sHvec[3 * HID + t] = seat_emb[st * 8 + t];
    }
    __syncthreads();

    const int MLP_IN = 3 * HID + 8; // 392
    if (tid < HID) {
        float acc = mlp_b1[tid];
        for (int j = 0; j < MLP_IN; ++j) acc += sHvec[j] * mlp_W1[j * HID + tid];
        sZ1[tid] = fmaxf(acc, 0.f);
    } else if (tid < HID + 64) {
        int t = tid - HID;
        float acc = rank_b1[t];
        for (int j = 0; j < MLP_IN; ++j) acc += sHvec[j] * rank_W1[j * 64 + t];
        sR[t] = fmaxf(acc, 0.f);
    }
    __syncthreads();

    if (tid < 64) {
        float acc = mlp_b2[tid];
        for (int j = 0; j < HID; ++j) acc += sZ1[j] * mlp_W2[j * 64 + tid];
        sZ2[tid] = fmaxf(acc, 0.f);
    }
    __syncthreads();

    float p = 0.f;
    if (tid < 64)       p = sZ2[tid] * mlp_W3[tid];
    else if (tid < 128) p = sR[tid - 64] * rank_W2[tid - 64];
#pragma unroll
    for (int off = 16; off > 0; off >>= 1) p += __shfl_xor(p, off, 32);
    if (lane == 0 && wave < 4) sRed[wave] = p;
    __syncthreads();

    if (tid == 0) {
        out[n]           = sRed[0] + sRed[1] + mlp_b3[0];
        float rk         = sRed[2] + sRed[3] + rank_b2[0];
        out[n_total + n] = 1.f / (1.f + expf(-rk));
    }
}

extern "C" void kernel_launch(void* const* d_in, const int* in_sizes, int n_in,
                              void* d_out, int out_size, void* d_ws, size_t ws_size,
                              hipStream_t stream) {
    (void)n_in; (void)d_ws; (void)ws_size; (void)out_size;
    const float* X       = (const float*)d_in[0];
    const float* A_hat   = (const float*)d_in[1];
    const int*   v1_idx  = (const int*)  d_in[2];
    const int*   v2_idx  = (const int*)  d_in[3];
    const int*   seat    = (const int*)  d_in[4];
    const float* conv0_W = (const float*)d_in[5];
    const float* conv0_b = (const float*)d_in[6];
    const float* ln0_g   = (const float*)d_in[7];
    const float* ln0_b   = (const float*)d_in[8];
    const float* conv1_W = (const float*)d_in[9];
    const float* conv1_b = (const float*)d_in[10];
    const float* ln1_g   = (const float*)d_in[11];
    const float* ln1_b   = (const float*)d_in[12];
    const float* conv2_W = (const float*)d_in[13];
    const float* conv2_b = (const float*)d_in[14];
    const float* ln2_g   = (const float*)d_in[15];
    const float* ln2_b   = (const float*)d_in[16];
    const float* s_emb   = (const float*)d_in[17];
    const float* mlp_W1  = (const float*)d_in[18];
    const float* mlp_b1  = (const float*)d_in[19];
    const float* mlp_W2  = (const float*)d_in[20];
    const float* mlp_b2  = (const float*)d_in[21];
    const float* mlp_W3  = (const float*)d_in[22];
    const float* mlp_b3  = (const float*)d_in[23];
    const float* rank_W1 = (const float*)d_in[24];
    const float* rank_b1 = (const float*)d_in[25];
    const float* rank_W2 = (const float*)d_in[26];
    const float* rank_b2 = (const float*)d_in[27];

    const int n_total = in_sizes[2];   // N = 16384 (length of v1_idx)

    gcn_fused_kernel<<<dim3(n_total), dim3(256), 0, stream>>>(
        X, A_hat, v1_idx, v2_idx, seat,
        conv0_W, conv0_b, ln0_g, ln0_b,
        conv1_W, conv1_b, ln1_g, ln1_b,
        conv2_W, conv2_b, ln2_g, ln2_b,
        s_emb,
        mlp_W1, mlp_b1, mlp_W2, mlp_b2, mlp_W3, mlp_b3,
        rank_W1, rank_b1, rank_W2, rank_b2,
        (float*)d_out, n_total);
}
